// PNATower_29368986370542
// MI455X (gfx1250) — matrix operand order repaired and checked by reading, not compiled, not run on hardware
//
#include <hip/hip_runtime.h>

typedef __attribute__((ext_vector_type(16))) __bf16 v16bf;
typedef __attribute__((ext_vector_type(8)))  float  v8f;

#define N_NODES 50000
#define N_EDGES 800000
#define AVG_D_LOG 2.8332f

union FragBF { v16bf v; uint4 q[2]; };

// Order-preserving float <-> u32 encoding so max/min lower to native
// global_atomic_max_u32 / global_atomic_min_u32 (no CAS loops).
__device__ __forceinline__ unsigned fenc(float f) {
  unsigned u = __float_as_uint(f);
  return (u & 0x80000000u) ? ~u : (u | 0x80000000u);
}
__device__ __forceinline__ float fdec(unsigned e) {
  unsigned u = (e & 0x80000000u) ? (e & 0x7fffffffu) : ~e;
  return __uint_as_float(u);
}

// ---------------- prep kernels ----------------

__global__ void k_cast_bf16(const float* __restrict__ s, __bf16* __restrict__ d, int n) {
  int i = blockIdx.x * blockDim.x + threadIdx.x;
  if (i < n) d[i] = (__bf16)s[i];
}

// W [K,64] fp32 row-major -> WT [64,K] bf16 (contiguous K per output column,
// so WMMA B fragments are two contiguous 16B loads per lane)
__global__ void k_transpose_w(const float* __restrict__ W, __bf16* __restrict__ WT, int K) {
  int i = blockIdx.x * blockDim.x + threadIdx.x;
  if (i >= K * 64) return;
  int c = i / K, k = i % K;
  WT[i] = (__bf16)W[(size_t)k * 64 + c];
}

__global__ void k_fill_u32(unsigned* __restrict__ p, unsigned v, int n) {
  int i = blockIdx.x * blockDim.x + threadIdx.x;
  if (i < n) p[i] = v;
}

__global__ void k_degree(const int* __restrict__ dst, float* __restrict__ deg) {
  int i = blockIdx.x * blockDim.x + threadIdx.x;
  if (i < N_EDGES) atomicAdd(&deg[dst[i]], 1.0f);
}

// ---------------- edge pretrans GEMM + fused scatter-aggregate ----------------
// z = [x[src] | x[dst] | e] (K wide) @ WT^T -> [16 edges x 16 cols] per wave.
// Epilogue: atomic sum/sumsq/max/min into per-node accumulators (live in L2).
template<int DIM1, int K>
__global__ __launch_bounds__(256)
void k_edge_gemm(const __bf16* __restrict__ xb, const __bf16* __restrict__ eb,
                 const __bf16* __restrict__ WT, const float* __restrict__ bias,
                 const int* __restrict__ src, const int* __restrict__ dst,
                 float* __restrict__ sum, float* __restrict__ sumsq,
                 unsigned* __restrict__ mxe, unsigned* __restrict__ mne)
{
  const int tid  = threadIdx.x;
  const int w    = tid >> 5, lane = tid & 31;
  const int mt   = w >> 2,   nt   = w & 3;
  const int m    = lane & 15, khalf = lane >> 4;
  const int ebase = blockIdx.x * 32 + mt * 16;   // E = 25000*32 exactly
  const int edge  = ebase + m;

  const int s = src[edge], d = dst[edge];
  const __bf16* rs = xb + (size_t)s * DIM1;
  const __bf16* rd = xb + (size_t)d * DIM1;
  const __bf16* re = eb + (size_t)edge * 32;

  const int col = nt * 16 + (lane & 15);
  const __bf16* wrow = WT + (size_t)col * K;

  v8f acc = {0.f, 0.f, 0.f, 0.f, 0.f, 0.f, 0.f, 0.f};
#pragma unroll
  for (int kb = 0; kb < K / 32; ++kb) {
    const int kbase = kb * 32;
    const __bf16* abase;
    if (kbase < DIM1)            abase = rs + kbase;            // x[src] span
    else if (kbase < 2 * DIM1)   abase = rd + (kbase - DIM1);   // x[dst] span
    else                         abase = re + (kbase - 2*DIM1); // edge span
    FragBF a, b;
    // A fragment, contiguous K order chosen here (same as B): lane-half holds K run [khalf*16..+15]
    a.q[0] = *(const uint4*)(abase + khalf * 16);
    a.q[1] = *(const uint4*)(abase + khalf * 16 + 8);
    // B fragment, same contiguous K order as A: lane-half holds K run of 16
    const __bf16* bbase = wrow + kbase + khalf * 16;
    b.q[0] = *(const uint4*)(bbase);
    b.q[1] = *(const uint4*)(bbase + 8);
    acc = __builtin_amdgcn_wmma_f32_16x16x32_bf16(false, a.v, false, b.v,
                                                  (short)0, acc, false, false);
  }

  const float bc = bias[col];
#pragma unroll
  for (int v = 0; v < 8; ++v) {
    const int e2 = ebase + khalf * 8 + v;          // C/D layout: M = v + 8*khalf
    const float val = acc[v] + bc;
    const size_t off = (size_t)dst[e2] * 64 + col;
    atomicAdd(&sum[off], val);
    atomicAdd(&sumsq[off], val * val);
    atomicMax(&mxe[off], fenc(val));
    atomicMin(&mne[off], fenc(val));
  }
}

// ---------------- node posttrans GEMM (PNA features built in LDS) ----------------
template<int DIM1, int K, bool IS_H>
__global__ __launch_bounds__(128)
void k_node_gemm(const __bf16* __restrict__ xb,
                 const float* __restrict__ sum, const float* __restrict__ sumsq,
                 const unsigned* __restrict__ mxe, const unsigned* __restrict__ mne,
                 const float* __restrict__ deg,
                 const __bf16* __restrict__ WT, const float* __restrict__ bias,
                 const float* __restrict__ snorm, float* __restrict__ out,
                 float* __restrict__ bnstats)
{
  __shared__ __bf16 feat[16 * K];
  __shared__ float bnloc[128];
  const int tid = threadIdx.x;
  const int nbase = blockIdx.x * 16;               // N = 3125*16 exactly

  if (IS_H) bnloc[tid] = 0.f;

  // stage node features: [ x (DIM1) | mean mx mn std | *amp | *att ] as bf16
  for (int i = tid; i < 16 * DIM1; i += 128) {
    int t = i / DIM1, j = i % DIM1;
    feat[t * K + j] = xb[(size_t)(nbase + t) * DIM1 + j];
  }
  for (int i = tid; i < 16 * 64; i += 128) {
    int t = i >> 6, c = i & 63;
    const int node = nbase + t;
    const size_t off = (size_t)node * 64 + c;
    const float dg = deg[node];
    const float dc = fmaxf(dg, 1.0f);
    const float mean = sum[off] / dc;
    const float sq   = sumsq[off] / dc;
    const float var  = fmaxf(sq - mean * mean, 0.0f);
    const float sd   = sqrtf(var + 1e-5f);
    const bool  has  = dg > 0.0f;
    const float mx = has ? fdec(mxe[off]) : 0.0f;
    const float mn = has ? fdec(mne[off]) : 0.0f;
    const float logd = has ? logf(dg + 1.0f) : 1.0f;
    const float amp = logd / AVG_D_LOG;
    const float att = AVG_D_LOG / logd;
    __bf16* fr = &feat[t * K + DIM1 + c];
    fr[0]   = (__bf16)mean;         fr[64]  = (__bf16)mx;
    fr[128] = (__bf16)mn;           fr[192] = (__bf16)sd;
    fr[256] = (__bf16)(mean * amp); fr[320] = (__bf16)(mx * amp);
    fr[384] = (__bf16)(mn * amp);   fr[448] = (__bf16)(sd * amp);
    fr[512] = (__bf16)(mean * att); fr[576] = (__bf16)(mx * att);
    fr[640] = (__bf16)(mn * att);   fr[704] = (__bf16)(sd * att);
  }
  __syncthreads();

  const int w = tid >> 5, lane = tid & 31;
  const int m = lane & 15, khalf = lane >> 4;
  const int col = w * 16 + m;
  const __bf16* wrow = WT + (size_t)col * K;

  v8f acc = {0.f, 0.f, 0.f, 0.f, 0.f, 0.f, 0.f, 0.f};
#pragma unroll
  for (int kb = 0; kb < K / 32; ++kb) {
    FragBF a, b;
    const __bf16* abase = &feat[m * K + kb * 32];
    a.q[0] = *(const uint4*)(abase + khalf * 16);
    a.q[1] = *(const uint4*)(abase + khalf * 16 + 8);
    const __bf16* bbase = wrow + kb * 32 + khalf * 16;
    b.q[0] = *(const uint4*)(bbase);
    b.q[1] = *(const uint4*)(bbase + 8);
    acc = __builtin_amdgcn_wmma_f32_16x16x32_bf16(false, a.v, false, b.v,
                                                  (short)0, acc, false, false);
  }

  const float bc = bias[col];
#pragma unroll
  for (int v = 0; v < 8; ++v) {
    const int node = nbase + khalf * 8 + v;
    float val = acc[v] + bc;
    if (IS_H) {
      val *= snorm[node];
      out[(size_t)node * 64 + col] = val;
      atomicAdd(&bnloc[col], val);            // LDS reduction first
      atomicAdd(&bnloc[64 + col], val * val);
    } else {
      out[(size_t)node * 64 + col] = val;
    }
  }
  if (IS_H) {
    __syncthreads();
    atomicAdd(&bnstats[tid], bnloc[tid]);     // 128 global atomics per WG
  }
}

// ---------------- batch-norm finalize (in place on h_out) ----------------
__global__ void k_bn(float* __restrict__ hout, const float* __restrict__ bnstats,
                     const float* __restrict__ gamma, const float* __restrict__ beta) {
  int i = blockIdx.x * blockDim.x + threadIdx.x;
  if (i >= N_NODES * 64) return;
  int c = i & 63;
  float mu  = bnstats[c] * (1.0f / N_NODES);
  float var = bnstats[64 + c] * (1.0f / N_NODES) - mu * mu;
  float inv = rsqrtf(var + 1e-5f);
  hout[i] = (hout[i] - mu) * inv * gamma[c] + beta[c];
}

// ---------------- launch ----------------
extern "C" void kernel_launch(void* const* d_in, const int* in_sizes, int n_in,
                              void* d_out, int out_size, void* d_ws, size_t ws_size,
                              hipStream_t stream) {
  (void)in_sizes; (void)n_in; (void)out_size; (void)ws_size;
  const float* h        = (const float*)d_in[0];
  const float* p        = (const float*)d_in[1];
  const float* e        = (const float*)d_in[2];
  const int*   src      = (const int*)d_in[3];
  const int*   dst      = (const int*)d_in[4];
  const float* snorm    = (const float*)d_in[5];
  const float* W_pre_h  = (const float*)d_in[6];
  const float* b_pre_h  = (const float*)d_in[7];
  const float* W_pre_p  = (const float*)d_in[8];
  const float* b_pre_p  = (const float*)d_in[9];
  const float* W_post_h = (const float*)d_in[10];
  const float* b_post_h = (const float*)d_in[11];
  const float* W_post_p = (const float*)d_in[12];
  const float* b_post_p = (const float*)d_in[13];
  const float* gamma    = (const float*)d_in[14];
  const float* beta     = (const float*)d_in[15];
  float* out = (float*)d_out;

  // workspace carve-up (256B aligned)
  char* ws = (char*)d_ws;
  size_t off = 0;
  auto alloc = [&](size_t bytes) -> void* {
    void* ptr = ws + off;
    off = (off + bytes + 255) & ~(size_t)255;
    return ptr;
  };
  const size_t NC = (size_t)N_NODES * 64;
  __bf16* hb        = (__bf16*)alloc((size_t)N_NODES * 128 * 2);
  __bf16* pb        = (__bf16*)alloc((size_t)N_NODES * 64 * 2);
  __bf16* ebuf      = (__bf16*)alloc((size_t)N_EDGES * 32 * 2);
  __bf16* WT_pre_h  = (__bf16*)alloc(64 * 288 * 2);
  __bf16* WT_pre_p  = (__bf16*)alloc(64 * 160 * 2);
  __bf16* WT_post_h = (__bf16*)alloc(64 * 896 * 2);
  __bf16* WT_post_p = (__bf16*)alloc(64 * 832 * 2);
  float* zblock = (float*)alloc(sizeof(float) * (6 * NC + N_NODES + 128));
  float*    sum_h   = zblock;
  float*    sumsq_h = zblock + NC;
  float*    sum_p   = zblock + 2 * NC;
  float*    sumsq_p = zblock + 3 * NC;
  unsigned* mxe_h   = (unsigned*)(zblock + 4 * NC);
  unsigned* mxe_p   = (unsigned*)(zblock + 5 * NC);
  float*    deg     = zblock + 6 * NC;
  float*    bnstats = zblock + 6 * NC + N_NODES;
  unsigned* mblock  = (unsigned*)alloc(sizeof(unsigned) * 2 * NC);
  unsigned* mne_h = mblock;
  unsigned* mne_p = mblock + NC;

  const int T = 256;
  // bf16 casts (halves gather traffic, enables bf16 WMMA)
  k_cast_bf16<<<(N_NODES * 128 + T - 1) / T, T, 0, stream>>>(h, hb, N_NODES * 128);
  k_cast_bf16<<<(N_NODES * 64  + T - 1) / T, T, 0, stream>>>(p, pb, N_NODES * 64);
  k_cast_bf16<<<(N_EDGES * 32  + T - 1) / T, T, 0, stream>>>(e, ebuf, N_EDGES * 32);
  // weight transposes to bf16 [64,K]
  k_transpose_w<<<(288 * 64 + T - 1) / T, T, 0, stream>>>(W_pre_h,  WT_pre_h,  288);
  k_transpose_w<<<(160 * 64 + T - 1) / T, T, 0, stream>>>(W_pre_p,  WT_pre_p,  160);
  k_transpose_w<<<(896 * 64 + T - 1) / T, T, 0, stream>>>(W_post_h, WT_post_h, 896);
  k_transpose_w<<<(832 * 64 + T - 1) / T, T, 0, stream>>>(W_post_p, WT_post_p, 832);
  // accumulator init: zeros (sums, sumsq, max-encodings, deg, bn) + min-encodings
  int zn = (int)(6 * NC + N_NODES + 128);
  k_fill_u32<<<(zn + T - 1) / T, T, 0, stream>>>((unsigned*)zblock, 0u, zn);
  k_fill_u32<<<((int)(2 * NC) + T - 1) / T, T, 0, stream>>>(mblock, 0xFFFFFFFFu, (int)(2 * NC));
  // degrees
  k_degree<<<(N_EDGES + T - 1) / T, T, 0, stream>>>(dst, deg);
  // edge pretrans GEMMs fused with scatter aggregation (accumulators live in L2)
  k_edge_gemm<128, 288><<<N_EDGES / 32, 256, 0, stream>>>(
      hb, ebuf, WT_pre_h, b_pre_h, src, dst, sum_h, sumsq_h, mxe_h, mne_h);
  k_edge_gemm<64, 160><<<N_EDGES / 32, 256, 0, stream>>>(
      pb, ebuf, WT_pre_p, b_pre_p, src, dst, sum_p, sumsq_p, mxe_p, mne_p);
  // node posttrans GEMMs (PNA feature build in LDS)
  k_node_gemm<128, 896, true><<<N_NODES / 16, 128, 0, stream>>>(
      hb, sum_h, sumsq_h, mxe_h, mne_h, deg, WT_post_h, b_post_h, snorm, out, bnstats);
  k_node_gemm<64, 832, false><<<N_NODES / 16, 128, 0, stream>>>(
      pb, sum_p, sumsq_p, mxe_p, mne_p, deg, WT_post_p, b_post_p, snorm,
      out + (size_t)N_NODES * 64, bnstats);
  // batch-norm finalize on h_out
  k_bn<<<(N_NODES * 64 + T - 1) / T, T, 0, stream>>>(out, bnstats, gamma, beta);
}
